// MultiHeadAttention_2808908611654
// MI455X (gfx1250) — matrix operand order repaired and checked by reading, not compiled, not run on hardware
//
#include <hip/hip_runtime.h>
#include <math.h>

// MultiHeadAttention forward for MI455X (gfx1250), wave32 + WMMA + async/TDM.
// B=2, S=2048, D=1024, H=16, DH=64.
// d_out = [out (B,S,D) f32 | weights (B,H,S,S) f32]

typedef __attribute__((ext_vector_type(16))) _Float16 v16h;
typedef __attribute__((ext_vector_type(8)))  float    v8f;
typedef float        f32x4 __attribute__((ext_vector_type(4)));
typedef unsigned int u32x4 __attribute__((ext_vector_type(4)));
typedef int          i32x8 __attribute__((ext_vector_type(8)));
typedef int          i32x4 __attribute__((ext_vector_type(4)));

#define B_    2
#define S_    2048
#define D_    1024
#define H_    16
#define DH_   64
#define NROWS 4096   // B*S

// Flat pointers into __shared__ live in the LDS aperture; the LDS offset is the low 32 address bits.
static __device__ __forceinline__ unsigned lds_off_of(const void* p) {
  return (unsigned)(unsigned long long)p;
}

// ---- async global -> LDS (16B per lane), tracked by ASYNCcnt ----
static __device__ __forceinline__ void async_copy_b128(unsigned lds_off, const void* g) {
  unsigned long long ga = (unsigned long long)g;
  asm volatile("global_load_async_to_lds_b128 %0, %1, off"
               :: "v"(lds_off), "v"(ga) : "memory");
}
static __device__ __forceinline__ void async_wait0() {
  asm volatile("s_wait_asynccnt 0x0" ::: "memory");
}
static __device__ __forceinline__ void async_wait_le4() {
  asm volatile("s_wait_asynccnt 0x4" ::: "memory");   // current tile (4 older ops) done
}

// ---- Tensor Data Mover: 2D tile of 2-byte elements, global -> LDS ----
// amdgpu-toolchain (clang-23) 6-arg builtin form.
static __device__ __forceinline__ void tdm_load_2d_f16(unsigned lds_byte_addr,
                                                       const void* gptr,
                                                       unsigned tile_w, unsigned tile_h,
                                                       unsigned stride_elems) {
  const unsigned long long ga = (unsigned long long)gptr;
  u32x4 g0;
  g0[0] = 1u;                                   // count=1, user descriptor
  g0[1] = lds_byte_addr;                        // lds_addr
  g0[2] = (unsigned)ga;                         // global_addr[31:0]
  g0[3] = (unsigned)(ga >> 32) | (2u << 30);    // global_addr[56:32] | type=2
  i32x8 g1;
  g1[0] = (int)(1u << 16);                      // workgroup_mask=0, data_size=1 (2B)
  g1[1] = (int)((tile_w & 0xFFFFu) << 16);      // tensor_dim0[15:0] @ bits63:48
  g1[2] = (int)(((tile_w >> 16) & 0xFFFFu) | ((tile_h & 0xFFFFu) << 16)); // dim0 hi | dim1 lo
  g1[3] = (int)(((tile_h >> 16) & 0xFFFFu) | ((tile_w & 0xFFFFu) << 16)); // dim1 hi | tile_dim0
  g1[4] = (int)(tile_h & 0xFFFFu);              // tile_dim1 | tile_dim2=0
  g1[5] = (int)stride_elems;                    // tensor_dim0_stride[31:0]
  g1[6] = 0;                                    // stride0 hi | stride1 lo
  g1[7] = 0;                                    // stride1 hi
  const i32x4 z4 = {0, 0, 0, 0};
  const i32x8 z8 = {0, 0, 0, 0, 0, 0, 0, 0};
  __builtin_amdgcn_tensor_load_to_lds(g0, g1, z4, z4, z8, 0);
}

// ---- WMMA fragment loads from LDS (contiguous k = 16*(lane/16) + s, same order in A and B) ----
static __device__ __forceinline__ v16h frag_a_lds(const _Float16* base, int ld,
                                                  int mbase, int lane) {
  v16h a;
  const _Float16* p = base + (mbase + (lane & 15)) * ld;
  const int koff = (lane & 16) ? 16 : 0;
#pragma unroll
  for (int j = 0; j < 8; ++j) {
    const int kb = 2 * j + koff;
    a[2 * j]     = p[kb];
    a[2 * j + 1] = p[kb + 1];
  }
  return a;
}
static __device__ __forceinline__ v16h frag_b_lds(const _Float16* base, int ld,
                                                  int nbase, int lane) {
  v16h b;
  const _Float16* p = base + (nbase + (lane & 15)) * ld + ((lane & 16) ? 16 : 0);
#pragma unroll
  for (int j = 0; j < 16; ++j) b[j] = p[j];
  return b;
}

#define WMMA_F16(a, b, c) \
  __builtin_amdgcn_wmma_f32_16x16x32_f16(false, (a), false, (b), (short)0, (c), false, false)

// =====================================================================
// Elementwise f32 -> f16 convert (inputs and weight matrices, done once).
// =====================================================================
__global__ __launch_bounds__(256) void cvt_f32_f16_kernel(const float* __restrict__ src,
                                                          _Float16* __restrict__ dst) {
  const size_t i = ((size_t)blockIdx.x * 256 + threadIdx.x) * 4;
  const f32x4 v = __builtin_nontemporal_load((const f32x4*)(src + i));
  dst[i + 0] = (_Float16)v.x;
  dst[i + 1] = (_Float16)v.y;
  dst[i + 2] = (_Float16)v.z;
  dst[i + 3] = (_Float16)v.w;
}

// =====================================================================
// Y[n,m] = sum_k X[n,k]*W[m,k] + bias[m], X/W already f16.
// Double-buffered LDS tiles streamed via global_load_async_to_lds_b128:
// issue tile k+1, wait ASYNCcnt<=4 (tile k complete), barrier, WMMA.
// Block 256 thr / 8 waves, tile 128x128, K-step 32, wave = 64x32.
// =====================================================================
template <bool OUT16>
__global__ __launch_bounds__(256) void gemm_f16_kernel(
    const _Float16* __restrict__ X, const _Float16* __restrict__ W,
    const float* __restrict__ bias, void* __restrict__ Yp) {
  __shared__ _Float16 As[2][128 * 32];
  __shared__ _Float16 Bs[2][128 * 32];
  const int tid = threadIdx.x, lane = tid & 31, wave = tid >> 5;
  const int waveM = wave & 1, waveN = wave >> 1;
  const int bm = blockIdx.x, bn = blockIdx.y;

  const v8f vzero = {0.f, 0.f, 0.f, 0.f, 0.f, 0.f, 0.f, 0.f};
  v8f acc[4][2];
#pragma unroll
  for (int i = 0; i < 4; ++i)
#pragma unroll
    for (int j = 0; j < 2; ++j) acc[i][j] = vzero;

  const int ldr = tid >> 1;          // 0..127 tile row
  const int ldh = (tid & 1) * 16;    // 0/16 half-of-row
  const _Float16* Xp = X + (size_t)(bm * 128 + ldr) * D_ + ldh;
  const _Float16* Wp = W + (size_t)(bn * 128 + ldr) * D_ + ldh;
  const unsigned toff = (unsigned)(ldr * 32 + ldh) * 2;
  unsigned aoff[2] = {lds_off_of(As[0]) + toff, lds_off_of(As[1]) + toff};
  unsigned boff[2] = {lds_off_of(Bs[0]) + toff, lds_off_of(Bs[1]) + toff};

  // prologue: stage first tile into buffer 0
  async_copy_b128(aoff[0],      Xp);
  async_copy_b128(aoff[0] + 16, Xp + 8);
  async_copy_b128(boff[0],      Wp);
  async_copy_b128(boff[0] + 16, Wp + 8);

  for (int k0 = 0; k0 < D_; k0 += 32) {
    const int cur = (k0 >> 5) & 1;
    const bool more = (k0 + 32) < D_;
    if (more) {                        // stage next tile into other buffer
      const int nxt = cur ^ 1;
      async_copy_b128(aoff[nxt],      Xp + k0 + 32);
      async_copy_b128(aoff[nxt] + 16, Xp + k0 + 40);
      async_copy_b128(boff[nxt],      Wp + k0 + 32);
      async_copy_b128(boff[nxt] + 16, Wp + k0 + 40);
      async_wait_le4();
    } else {
      async_wait0();
    }
    __syncthreads();
    const v16h bf0 = frag_b_lds(Bs[cur], 32, waveN * 32 + 0, lane);
    const v16h bf1 = frag_b_lds(Bs[cur], 32, waveN * 32 + 16, lane);
#pragma unroll
    for (int mt = 0; mt < 4; ++mt) {
      const v16h af = frag_a_lds(As[cur], 32, waveM * 64 + mt * 16, lane);
      acc[mt][0] = WMMA_F16(af, bf0, acc[mt][0]);
      acc[mt][1] = WMMA_F16(af, bf1, acc[mt][1]);
    }
    __syncthreads();
  }

  const int rofs = (lane & 16) ? 8 : 0;
  const int cin = lane & 15;
#pragma unroll
  for (int mt = 0; mt < 4; ++mt)
#pragma unroll
    for (int nt = 0; nt < 2; ++nt) {
      const int col = bn * 128 + waveN * 32 + nt * 16 + cin;
      const float bv = bias[col];
#pragma unroll
      for (int r = 0; r < 8; ++r) {
        const int row = bm * 128 + waveM * 64 + mt * 16 + rofs + r;
        const float val = acc[mt][nt][r] + bv;
        const size_t o = (size_t)row * D_ + col;
        if (OUT16) ((_Float16*)Yp)[o] = (_Float16)val;
        else       ((float*)Yp)[o] = val;
      }
    }
}

// =====================================================================
// scores[bh,i,j] = 0.125*sum_{k<64} q[b,i,h,k]*kk[b,j,h,k] + mask[b,i,j]*(-1e30)
// Q/K tiles (128x32 f16, row stride 1024) fetched by the Tensor Data Mover,
// double buffered over the two K-steps; NT stores (pure streaming output).
// =====================================================================
__global__ __launch_bounds__(256) void attn_scores_kernel(
    const _Float16* __restrict__ qp, const _Float16* __restrict__ kp,
    const float* __restrict__ mask, float* __restrict__ wbuf) {
  __shared__ _Float16 As[2][128 * 32];
  __shared__ _Float16 Bs[2][128 * 32];
  const int tid = threadIdx.x, lane = tid & 31, wave = tid >> 5;
  const int waveM = wave & 1, waveN = wave >> 1;
  const int bm = blockIdx.x, bn = blockIdx.y, bh = blockIdx.z;
  const int b = bh >> 4, h = bh & 15;

  const v8f vzero = {0.f, 0.f, 0.f, 0.f, 0.f, 0.f, 0.f, 0.f};
  v8f acc[4][2];
#pragma unroll
  for (int i = 0; i < 4; ++i)
#pragma unroll
    for (int j = 0; j < 2; ++j) acc[i][j] = vzero;

  const _Float16* Qt = qp + (size_t)(b * S_ + bm * 128) * D_ + h * DH_;
  const _Float16* Kt = kp + (size_t)(b * S_ + bn * 128) * D_ + h * DH_;

  if (tid == 0) {                       // stage both K-steps via TDM, back-to-back
    tdm_load_2d_f16(lds_off_of(As[0]), Qt,      32, 128, D_);
    tdm_load_2d_f16(lds_off_of(Bs[0]), Kt,      32, 128, D_);
    tdm_load_2d_f16(lds_off_of(As[1]), Qt + 32, 32, 128, D_);
    tdm_load_2d_f16(lds_off_of(Bs[1]), Kt + 32, 32, 128, D_);
    __builtin_amdgcn_s_wait_tensorcnt(2);      // first pair complete
  }
  __syncthreads();

#pragma unroll
  for (int half = 0; half < 2; ++half) {
    const v16h bf0 = frag_b_lds(Bs[half], 32, waveN * 32 + 0, lane);
    const v16h bf1 = frag_b_lds(Bs[half], 32, waveN * 32 + 16, lane);
#pragma unroll
    for (int mt = 0; mt < 4; ++mt) {
      const v16h af = frag_a_lds(As[half], 32, waveM * 64 + mt * 16, lane);
      acc[mt][0] = WMMA_F16(af, bf0, acc[mt][0]);
      acc[mt][1] = WMMA_F16(af, bf1, acc[mt][1]);
    }
    if (half == 0) {
      if (tid == 0) __builtin_amdgcn_s_wait_tensorcnt(0);  // second pair complete
      __syncthreads();
    }
  }

  const int rofs = (lane & 16) ? 8 : 0;
  const int cin = lane & 15;
#pragma unroll
  for (int mt = 0; mt < 4; ++mt)
#pragma unroll
    for (int nt = 0; nt < 2; ++nt) {
      const int j = bn * 128 + waveN * 32 + nt * 16 + cin;
#pragma unroll
      for (int r = 0; r < 8; ++r) {
        const int i = bm * 128 + waveM * 64 + mt * 16 + rofs + r;
        const float mv = __builtin_nontemporal_load(&mask[((size_t)b * S_ + i) * S_ + j]);
        __builtin_nontemporal_store(acc[mt][nt][r] * 0.125f + mv * (-1e30f),
                                    &wbuf[((size_t)bh * S_ + i) * S_ + j]);
      }
    }
}

// =====================================================================
// In-place row softmax over 2048-wide rows (B*H*S rows). Streaming -> NT.
// =====================================================================
__global__ __launch_bounds__(256) void softmax_kernel(float* __restrict__ wbuf) {
  float* p = wbuf + (size_t)blockIdx.x * S_;
  const int tid = threadIdx.x;
  __shared__ float redm[8];
  __shared__ float reds[8];
  float v[8];
  float m = -3.0e38f;
#pragma unroll
  for (int u = 0; u < 8; ++u) {
    v[u] = __builtin_nontemporal_load(&p[tid * 8 + u]);
    m = fmaxf(m, v[u]);
  }
#pragma unroll
  for (int off = 16; off > 0; off >>= 1) m = fmaxf(m, __shfl_xor(m, off, 32));
  if ((tid & 31) == 0) redm[tid >> 5] = m;
  __syncthreads();
  float bm = redm[0];
#pragma unroll
  for (int i = 1; i < 8; ++i) bm = fmaxf(bm, redm[i]);
  float s = 0.f;
#pragma unroll
  for (int u = 0; u < 8; ++u) { v[u] = __expf(v[u] - bm); s += v[u]; }
#pragma unroll
  for (int off = 16; off > 0; off >>= 1) s += __shfl_xor(s, off, 32);
  if ((tid & 31) == 0) reds[tid >> 5] = s;
  __syncthreads();
  float ts = 0.f;
#pragma unroll
  for (int i = 0; i < 8; ++i) ts += reds[i];
  const float inv = 1.0f / ts;
#pragma unroll
  for (int u = 0; u < 8; ++u)
    __builtin_nontemporal_store(v[u] * inv, &p[tid * 8 + u]);
}

// =====================================================================
// attn[b,s,h*64+n] = sum_j weights[bh,s,j] * v[b,j,h*64+n]  (f16 output)
// Weights f32 (streaming, NT loads) -> VALU convert on A-tile stage.
// =====================================================================
__global__ __launch_bounds__(256) void attn_v_kernel(
    const float* __restrict__ wbuf, const _Float16* __restrict__ vp,
    _Float16* __restrict__ attn16) {
  __shared__ _Float16 As[128 * 32];
  __shared__ _Float16 Bs[64 * 32];   // Bs[n][k]
  const int tid = threadIdx.x, lane = tid & 31, wave = tid >> 5;
  const int bm = blockIdx.x, bh = blockIdx.y;
  const int b = bh >> 4, h = bh & 15;

  const v8f vzero = {0.f, 0.f, 0.f, 0.f, 0.f, 0.f, 0.f, 0.f};
  v8f acc[4];
#pragma unroll
  for (int i = 0; i < 4; ++i) acc[i] = vzero;

  const int ldr = tid >> 1, ldk = (tid & 1) * 16;
  const float* Wp = wbuf + ((size_t)bh * S_ + bm * 128 + ldr) * S_ + ldk;
  const int vk = tid >> 3;         // 0..31 : k within tile
  const int vn = (tid & 7) * 8;    // 0..56 : n start
  const _Float16* Vp = vp + (size_t)(b * S_ + vk) * D_ + h * DH_ + vn;

  for (int k0 = 0; k0 < S_; k0 += 32) {
#pragma unroll
    for (int u = 0; u < 16; u += 4) {
      const f32x4 wv = __builtin_nontemporal_load((const f32x4*)(Wp + k0 + u));
      _Float16* a = &As[ldr * 32 + ldk + u];
      a[0] = (_Float16)wv.x; a[1] = (_Float16)wv.y;
      a[2] = (_Float16)wv.z; a[3] = (_Float16)wv.w;
    }
    const _Float16* vsrc = Vp + (size_t)k0 * D_;
#pragma unroll
    for (int u = 0; u < 8; ++u) Bs[(vn + u) * 32 + vk] = vsrc[u];  // transpose to [n][k]
    if (k0 + 32 < S_) __builtin_prefetch(Wp + k0 + 32, 0, 3);
    __syncthreads();
    const v16h af = frag_a_lds(As, 32, wave * 16, lane);
#pragma unroll
    for (int nt = 0; nt < 4; ++nt) {
      const v16h bf = frag_b_lds(Bs, 32, nt * 16, lane);
      acc[nt] = WMMA_F16(af, bf, acc[nt]);
    }
    __syncthreads();
  }

  const int rofs = (lane & 16) ? 8 : 0;
  const int cin = lane & 15;
#pragma unroll
  for (int nt = 0; nt < 4; ++nt)
#pragma unroll
    for (int r = 0; r < 8; ++r) {
      const int srow = bm * 128 + wave * 16 + rofs + r;
      attn16[(size_t)(b * S_ + srow) * D_ + h * DH_ + nt * 16 + cin] = (_Float16)acc[nt][r];
    }
}

// =====================================================================
extern "C" void kernel_launch(void* const* d_in, const int* in_sizes, int n_in,
                              void* d_out, int out_size, void* d_ws, size_t ws_size,
                              hipStream_t stream) {
  const float* q    = (const float*)d_in[0];
  const float* k    = (const float*)d_in[1];
  const float* v    = (const float*)d_in[2];
  const float* mask = (const float*)d_in[3];
  const float* wq_w = (const float*)d_in[4];
  const float* wq_b = (const float*)d_in[5];
  const float* wk_w = (const float*)d_in[6];
  const float* wk_b = (const float*)d_in[7];
  const float* wv_w = (const float*)d_in[8];
  const float* wv_b = (const float*)d_in[9];
  const float* wo_w = (const float*)d_in[10];
  const float* wo_b = (const float*)d_in[11];

  float* out  = (float*)d_out;
  float* wbuf = out + (size_t)NROWS * D_;     // weights output region (B,H,S,S)

  // workspace (all f16): q16|k16|v16|qp16|kp16|vp16|attn16 (NROWS*D each) + 4 weight mats
  _Float16* q16  = (_Float16*)d_ws;
  _Float16* k16  = q16  + (size_t)NROWS * D_;
  _Float16* v16  = k16  + (size_t)NROWS * D_;
  _Float16* qp16 = v16  + (size_t)NROWS * D_;
  _Float16* kp16 = qp16 + (size_t)NROWS * D_;
  _Float16* vp16 = kp16 + (size_t)NROWS * D_;
  _Float16* at16 = vp16 + (size_t)NROWS * D_;
  _Float16* wq16 = at16 + (size_t)NROWS * D_;
  _Float16* wk16 = wq16 + (size_t)D_ * D_;
  _Float16* wv16 = wk16 + (size_t)D_ * D_;
  _Float16* wo16 = wv16 + (size_t)D_ * D_;

  const dim3 blk(256);
  const int gBig = NROWS * D_ / (256 * 4);   // 4096 blocks
  const int gWgt = D_ * D_ / (256 * 4);      // 1024 blocks

  cvt_f32_f16_kernel<<<gBig, blk, 0, stream>>>(q, q16);
  cvt_f32_f16_kernel<<<gBig, blk, 0, stream>>>(k, k16);
  cvt_f32_f16_kernel<<<gBig, blk, 0, stream>>>(v, v16);
  cvt_f32_f16_kernel<<<gWgt, blk, 0, stream>>>(wq_w, wq16);
  cvt_f32_f16_kernel<<<gWgt, blk, 0, stream>>>(wk_w, wk16);
  cvt_f32_f16_kernel<<<gWgt, blk, 0, stream>>>(wv_w, wv16);
  cvt_f32_f16_kernel<<<gWgt, blk, 0, stream>>>(wo_w, wo16);

  const dim3 gProj(NROWS / 128, D_ / 128);
  gemm_f16_kernel<true><<<gProj, blk, 0, stream>>>(q16, wq16, wq_b, qp16);
  gemm_f16_kernel<true><<<gProj, blk, 0, stream>>>(k16, wk16, wk_b, kp16);
  gemm_f16_kernel<true><<<gProj, blk, 0, stream>>>(v16, wv16, wv_b, vp16);

  const dim3 gSc(S_ / 128, S_ / 128, B_ * H_);
  attn_scores_kernel<<<gSc, blk, 0, stream>>>(qp16, kp16, mask, wbuf);

  softmax_kernel<<<dim3(B_ * H_ * S_), blk, 0, stream>>>(wbuf);

  const dim3 gAv(S_ / 128, B_ * H_);
  attn_v_kernel<<<gAv, blk, 0, stream>>>(wbuf, vp16, at16);

  gemm_f16_kernel<false><<<gProj, blk, 0, stream>>>(at16, wo16, wo_b, out);
}
